// MultiHeadAttention_87969520157109
// MI455X (gfx1250) — matrix operand order repaired and checked by reading, not compiled, not run on hardware
//
#include <hip/hip_runtime.h>
#include <hip/hip_bf16.h>
#include <math.h>

#define BDIM  2
#define SEQ   2048
#define DMODEL 1024
#define NHEAD 16
#define DKH   64

typedef __attribute__((ext_vector_type(16))) _Float16 v16h;
typedef __attribute__((ext_vector_type(8)))  float    v8f;

union FragH {
  v16h v;
  uint4 u4[2];
  _Float16 h[16];
};

// Empty asm register fence: forces the fragment to be materialized (all loads
// complete) before any instruction after the fence, and keeps it in its own
// VGPR tuple so the scheduler cannot re-serialize load->wmma pairs.
#define FRAG_FENCE(f) asm volatile("" : "+v"((f).v))

__device__ __forceinline__ v8f wmma_f16(const FragH& a, const FragH& b, v8f c) {
  // D = A(16x32 f16) * B(32x16 f16) + C(16x16 f32)
  return __builtin_amdgcn_wmma_f32_16x16x32_f16(false, a.v, false, b.v,
                                                (short)0, c, false, false);
}

// ---------------------------------------------------------------------------
// Kernel 1: QKV projections.  X[B*S, D] (fp32) @ W[D, D] (fp32) + b -> f16.
// mode (blockIdx.z): 0 -> Qh [B,H,S,DK], 1 -> Kh [B,H,S,DK], 2 -> Vt [B,H,DK,S]
// Block: 256 threads (8 waves). Block tile 128(M) x 64(N). K-step 32.
// Double-buffered LDS W tile: one barrier per k-step, staging overlaps WMMA.
// ---------------------------------------------------------------------------
__global__ __launch_bounds__(256)
void proj_qkv_kernel(const float* __restrict__ X0, const float* __restrict__ X1,
                     const float* __restrict__ X2,
                     const float* __restrict__ Wq, const float* __restrict__ bq,
                     const float* __restrict__ Wk, const float* __restrict__ bk,
                     const float* __restrict__ Wv, const float* __restrict__ bv,
                     _Float16* __restrict__ Qh, _Float16* __restrict__ Kh,
                     _Float16* __restrict__ Vt)
{
  const int mode = blockIdx.z;
  const float* X    = (mode == 0) ? X0 : ((mode == 1) ? X1 : X2);
  const float* W    = (mode == 0) ? Wq : ((mode == 1) ? Wk : Wv);
  const float* bias = (mode == 0) ? bq : ((mode == 1) ? bk : bv);

  __shared__ __align__(16) _Float16 wt[2][64 * 32];  // ping-pong [n][k] tiles

  const int tid  = threadIdx.x;
  const int wave = tid >> 5;
  const int lane = tid & 31;
  const int l16  = lane & 15;
  const int hb   = lane >> 4;           // lane half: 0 or 1
  const int mtile = blockIdx.x * 128 + wave * 16;
  const int mrow  = mtile + l16;        // A-fragment row for this lane
  const int nbase = blockIdx.y * 64;
  const int c4 = tid & 15;              // n in chunks of 4
  const int r0 = tid >> 4;              // k row 0..15

  auto stage = [&](int buf, int kk) {
#pragma unroll
    for (int rr = 0; rr < 2; ++rr) {
      const int r = r0 + rr * 16;
      const float4 w4 = *reinterpret_cast<const float4*>(
          &W[(size_t)(kk + r) * DMODEL + nbase + c4 * 4]);
      wt[buf][(c4 * 4 + 0) * 32 + r] = (_Float16)w4.x;
      wt[buf][(c4 * 4 + 1) * 32 + r] = (_Float16)w4.y;
      wt[buf][(c4 * 4 + 2) * 32 + r] = (_Float16)w4.z;
      wt[buf][(c4 * 4 + 3) * 32 + r] = (_Float16)w4.w;
    }
  };

  v8f acc[4] = {};
  stage(0, 0);

  const int NIT = DMODEL / 32;
  for (int it = 0; it < NIT; ++it) {
    __syncthreads();                       // staged tile `it & 1` visible
    if (it + 1 < NIT) stage((it + 1) & 1, (it + 1) * 32);

    const int kk = it * 32;
    // A fragment: row = mrow, halves 0..15 -> K=16*lanehalf+0..15
    FragH a;
    const float* ar = &X[(size_t)mrow * DMODEL + kk + hb * 16];
#pragma unroll
    for (int i = 0; i < 8; ++i) a.h[i]     = (_Float16)ar[i];
#pragma unroll
    for (int i = 0; i < 8; ++i) a.h[8 + i] = (_Float16)ar[8 + i];

    // Batch all B fragments into distinct live tuples, then WMMAs
    // back-to-back behind a single dscnt wait.
    FragH bf0, bf1, bf2, bf3;
    {
      const uint4* p0 = reinterpret_cast<const uint4*>(
          &wt[it & 1][(0 * 16 + l16) * 32 + hb * 16]);
      const uint4* p1 = reinterpret_cast<const uint4*>(
          &wt[it & 1][(1 * 16 + l16) * 32 + hb * 16]);
      const uint4* p2 = reinterpret_cast<const uint4*>(
          &wt[it & 1][(2 * 16 + l16) * 32 + hb * 16]);
      const uint4* p3 = reinterpret_cast<const uint4*>(
          &wt[it & 1][(3 * 16 + l16) * 32 + hb * 16]);
      bf0.u4[0] = p0[0]; bf0.u4[1] = p0[1];
      bf1.u4[0] = p1[0]; bf1.u4[1] = p1[1];
      bf2.u4[0] = p2[0]; bf2.u4[1] = p2[1];
      bf3.u4[0] = p3[0]; bf3.u4[1] = p3[1];
    }
    FRAG_FENCE(a);
    FRAG_FENCE(bf0); FRAG_FENCE(bf1); FRAG_FENCE(bf2); FRAG_FENCE(bf3);
    acc[0] = wmma_f16(a, bf0, acc[0]);
    acc[1] = wmma_f16(a, bf1, acc[1]);
    acc[2] = wmma_f16(a, bf2, acc[2]);
    acc[3] = wmma_f16(a, bf3, acc[3]);
  }

  // Epilogue: bias add, convert to f16, scatter to head-major layouts.
#pragma unroll
  for (int j = 0; j < 4; ++j) {
    const int ncol = nbase + j * 16 + l16;
    const float bv_ = bias[ncol];
    const int h   = ncol >> 6;    // / DKH
    const int dkc = ncol & 63;    // % DKH
#pragma unroll
    for (int r = 0; r < 8; ++r) {
      const int m  = mtile + r + hb * 8;   // C layout: M = r + 8*lanehalf
      const int b_ = m >> 11;              // / SEQ
      const int s  = m & (SEQ - 1);
      const float val = acc[j][r] + bv_;
      if (mode == 2) {
        Vt[(((size_t)b_ * NHEAD + h) * DKH + dkc) * SEQ + s] = (_Float16)val;
      } else {
        _Float16* dst = (mode == 0) ? Qh : Kh;
        dst[(((size_t)b_ * NHEAD + h) * SEQ + s) * DKH + dkc] = (_Float16)val;
      }
    }
  }
}

// ---------------------------------------------------------------------------
// Kernel 2: fused attention. One wave per 16-row q tile of one (b, h).
// Pass 1: per-lane online softmax stats, single cross-lane merge at the end.
// Pass 2: recompute logits, emit normalized weights, accumulate w @ V (WMMA).
// Block: 128 threads (4 waves).
// ---------------------------------------------------------------------------
__global__ __launch_bounds__(128)
void attention_kernel(const _Float16* __restrict__ Qh,
                      const _Float16* __restrict__ Kh,
                      const _Float16* __restrict__ Vt,
                      const int* __restrict__ mask,
                      float* __restrict__ weights,
                      _Float16* __restrict__ attn)
{
  __shared__ __align__(16) _Float16 wlds[4][16 * 32];

  const int tid  = threadIdx.x;
  const int wave = tid >> 5;
  const int lane = tid & 31;
  const int l16  = lane & 15;
  const int hb   = lane >> 4;
  const int h = blockIdx.y;
  const int b = blockIdx.z;
  const int qbase = (blockIdx.x * 4 + wave) * 16;
  const float scale = 0.125f;  // 1/sqrt(DK)

  const size_t bh = (size_t)b * NHEAD + h;
  const _Float16* Qp = Qh + bh * SEQ * DKH;
  const _Float16* Kp = Kh + bh * SEQ * DKH;
  const _Float16* Vp = Vt + bh * DKH * SEQ;
  const int* mp = mask + (size_t)b * SEQ * SEQ;
  float* wout = weights + bh * SEQ * SEQ;

  // Q A-fragments, hoisted: 2 k-steps over DK=64.
  FragH qa[2];
  {
    const _Float16* qrow = Qp + (size_t)(qbase + l16) * DKH;
#pragma unroll
    for (int s = 0; s < 2; ++s) {
      const uint4* p = reinterpret_cast<const uint4*>(qrow + s * 32 + hb * 16);
      qa[s].u4[0] = p[0];   // halves 0..7  -> K = koff..koff+7
      qa[s].u4[1] = p[1];   // halves 8..15 -> K = koff+8..koff+15
    }
  }

  auto logits_tile = [&](int kt) -> v8f {
    FragH kb0, kb1;  // B frags: N = kcol, Kdim = s*32 + 16*lanehalf + 0..15
    const uint4* p0 = reinterpret_cast<const uint4*>(
        Kp + (size_t)(kt * 16 + l16) * DKH + 0 * 32 + hb * 16);
    const uint4* p1 = reinterpret_cast<const uint4*>(
        Kp + (size_t)(kt * 16 + l16) * DKH + 1 * 32 + hb * 16);
    kb0.u4[0] = p0[0]; kb0.u4[1] = p0[1];
    kb1.u4[0] = p1[0]; kb1.u4[1] = p1[1];
    FRAG_FENCE(kb0); FRAG_FENCE(kb1);
    v8f c = {};
    c = wmma_f16(qa[0], kb0, c);
    c = wmma_f16(qa[1], kb1, c);
    return c;
  };

  // ---- Pass 1: per-lane online max / exp-sum over this lane's columns ----
  float mrow[8], lrow[8];
#pragma unroll
  for (int r = 0; r < 8; ++r) { mrow[r] = -3.0e38f; lrow[r] = 0.0f; }

  for (int kt = 0; kt < SEQ / 16; ++kt) {
    v8f c = logits_tile(kt);
#pragma unroll
    for (int r = 0; r < 8; ++r) {
      const int q = qbase + r + hb * 8;
      const float mk = (float)mp[(size_t)q * SEQ + kt * 16 + l16];
      const float lg = c[r] * scale - 1.0e9f * mk;
      const float mnew = fmaxf(mrow[r], lg);
      lrow[r] = lrow[r] * __expf(mrow[r] - mnew) + __expf(lg - mnew);
      mrow[r] = mnew;
    }
  }
  // Merge (m, l) pairs across the 16 lanes of each half (4 xor steps).
#pragma unroll
  for (int r = 0; r < 8; ++r) {
#pragma unroll
    for (int d = 1; d < 16; d <<= 1) {
      const float mo = __shfl_xor(mrow[r], d, 32);
      const float lo = __shfl_xor(lrow[r], d, 32);
      const float mn = fmaxf(mrow[r], mo);
      lrow[r] = lrow[r] * __expf(mrow[r] - mn) + lo * __expf(mo - mn);
      mrow[r] = mn;
    }
  }
  float rinv[8];
#pragma unroll
  for (int r = 0; r < 8; ++r) rinv[r] = 1.0f / lrow[r];

  // ---- Pass 2: write weights + accumulate attn = w @ V ----
  v8f acc[4] = {};
  for (int k2 = 0; k2 < SEQ / 32; ++k2) {
#pragma unroll
    for (int t = 0; t < 2; ++t) {
      const int kt = k2 * 2 + t;
      v8f c = logits_tile(kt);
#pragma unroll
      for (int r = 0; r < 8; ++r) {
        const int q = qbase + r + hb * 8;
        const int kcol = kt * 16 + l16;
        const float mk = (float)mp[(size_t)q * SEQ + kcol];
        const float lg = c[r] * scale - 1.0e9f * mk;
        const float w = __expf(lg - mrow[r]) * rinv[r];
        wout[(size_t)q * SEQ + kcol] = w;
        wlds[wave][(r + hb * 8) * 32 + t * 16 + l16] = (_Float16)w;
      }
    }
    __syncthreads();
    FragH wa;  // A-frag for w tile (16 x 32 over seq)
    {
      const uint4* p = reinterpret_cast<const uint4*>(
          &wlds[wave][l16 * 32 + hb * 16]);
      wa.u4[0] = p[0];
      wa.u4[1] = p[1];
    }
    __syncthreads();
    // Batch all V fragments into live tuples, then back-to-back WMMAs.
    FragH vb0, vb1, vb2, vb3;
    {
      const uint4* p0 = reinterpret_cast<const uint4*>(
          Vp + (size_t)(0 * 16 + l16) * SEQ + k2 * 32 + hb * 16);
      const uint4* p1 = reinterpret_cast<const uint4*>(
          Vp + (size_t)(1 * 16 + l16) * SEQ + k2 * 32 + hb * 16);
      const uint4* p2 = reinterpret_cast<const uint4*>(
          Vp + (size_t)(2 * 16 + l16) * SEQ + k2 * 32 + hb * 16);
      const uint4* p3 = reinterpret_cast<const uint4*>(
          Vp + (size_t)(3 * 16 + l16) * SEQ + k2 * 32 + hb * 16);
      vb0.u4[0] = p0[0]; vb0.u4[1] = p0[1];
      vb1.u4[0] = p1[0]; vb1.u4[1] = p1[1];
      vb2.u4[0] = p2[0]; vb2.u4[1] = p2[1];
      vb3.u4[0] = p3[0]; vb3.u4[1] = p3[1];
    }
    FRAG_FENCE(wa);
    FRAG_FENCE(vb0); FRAG_FENCE(vb1); FRAG_FENCE(vb2); FRAG_FENCE(vb3);
    acc[0] = wmma_f16(wa, vb0, acc[0]);
    acc[1] = wmma_f16(wa, vb1, acc[1]);
    acc[2] = wmma_f16(wa, vb2, acc[2]);
    acc[3] = wmma_f16(wa, vb3, acc[3]);
  }

  // Store attn (merged heads, f16) for the output projection.
#pragma unroll
  for (int n = 0; n < 4; ++n) {
#pragma unroll
    for (int r = 0; r < 8; ++r) {
      const int q = qbase + r + hb * 8;
      const size_t row = (size_t)b * SEQ + q;
      attn[row * DMODEL + h * DKH + n * 16 + l16] = (_Float16)acc[n][r];
    }
  }
}

// ---------------------------------------------------------------------------
// Kernel 3: output projection. attn[B*S, D] (f16) @ Wo (fp32->f16) + bo -> f32
// ---------------------------------------------------------------------------
__global__ __launch_bounds__(256)
void proj_out_kernel(const _Float16* __restrict__ A,
                     const float* __restrict__ Wo,
                     const float* __restrict__ bo,
                     float* __restrict__ out)
{
  __shared__ __align__(16) _Float16 wt[2][64 * 32];

  const int tid  = threadIdx.x;
  const int wave = tid >> 5;
  const int lane = tid & 31;
  const int l16  = lane & 15;
  const int hb   = lane >> 4;
  const int mtile = blockIdx.x * 128 + wave * 16;
  const int mrow  = mtile + l16;
  const int nbase = blockIdx.y * 64;
  const int c4 = tid & 15;
  const int r0 = tid >> 4;

  auto stage = [&](int buf, int kk) {
#pragma unroll
    for (int rr = 0; rr < 2; ++rr) {
      const int r = r0 + rr * 16;
      const float4 w4 = *reinterpret_cast<const float4*>(
          &Wo[(size_t)(kk + r) * DMODEL + nbase + c4 * 4]);
      wt[buf][(c4 * 4 + 0) * 32 + r] = (_Float16)w4.x;
      wt[buf][(c4 * 4 + 1) * 32 + r] = (_Float16)w4.y;
      wt[buf][(c4 * 4 + 2) * 32 + r] = (_Float16)w4.z;
      wt[buf][(c4 * 4 + 3) * 32 + r] = (_Float16)w4.w;
    }
  };

  v8f acc[4] = {};
  stage(0, 0);

  const int NIT = DMODEL / 32;
  for (int it = 0; it < NIT; ++it) {
    __syncthreads();
    if (it + 1 < NIT) stage((it + 1) & 1, (it + 1) * 32);

    const int kk = it * 32;
    FragH a;
    const uint4* ap = reinterpret_cast<const uint4*>(
        &A[(size_t)mrow * DMODEL + kk + hb * 16]);
    a.u4[0] = ap[0];
    a.u4[1] = ap[1];

    FragH bf0, bf1, bf2, bf3;
    {
      const uint4* p0 = reinterpret_cast<const uint4*>(
          &wt[it & 1][(0 * 16 + l16) * 32 + hb * 16]);
      const uint4* p1 = reinterpret_cast<const uint4*>(
          &wt[it & 1][(1 * 16 + l16) * 32 + hb * 16]);
      const uint4* p2 = reinterpret_cast<const uint4*>(
          &wt[it & 1][(2 * 16 + l16) * 32 + hb * 16]);
      const uint4* p3 = reinterpret_cast<const uint4*>(
          &wt[it & 1][(3 * 16 + l16) * 32 + hb * 16]);
      bf0.u4[0] = p0[0]; bf0.u4[1] = p0[1];
      bf1.u4[0] = p1[0]; bf1.u4[1] = p1[1];
      bf2.u4[0] = p2[0]; bf2.u4[1] = p2[1];
      bf3.u4[0] = p3[0]; bf3.u4[1] = p3[1];
    }
    FRAG_FENCE(a);
    FRAG_FENCE(bf0); FRAG_FENCE(bf1); FRAG_FENCE(bf2); FRAG_FENCE(bf3);
    acc[0] = wmma_f16(a, bf0, acc[0]);
    acc[1] = wmma_f16(a, bf1, acc[1]);
    acc[2] = wmma_f16(a, bf2, acc[2]);
    acc[3] = wmma_f16(a, bf3, acc[3]);
  }

#pragma unroll
  for (int j = 0; j < 4; ++j) {
    const int ncol = nbase + j * 16 + l16;
    const float bv_ = bo[ncol];
#pragma unroll
    for (int r = 0; r < 8; ++r) {
      const int m = mtile + r + hb * 8;
      out[(size_t)m * DMODEL + ncol] = acc[j][r] + bv_;
    }
  }
}

// ---------------------------------------------------------------------------
extern "C" void kernel_launch(void* const* d_in, const int* in_sizes, int n_in,
                              void* d_out, int out_size, void* d_ws, size_t ws_size,
                              hipStream_t stream) {
  const float* query = (const float*)d_in[0];
  const float* key   = (const float*)d_in[1];
  const float* value = (const float*)d_in[2];
  const int*   mask  = (const int*)d_in[3];
  const float* Wq = (const float*)d_in[4];
  const float* bq = (const float*)d_in[5];
  const float* Wk = (const float*)d_in[6];
  const float* bk = (const float*)d_in[7];
  const float* Wv = (const float*)d_in[8];
  const float* bv = (const float*)d_in[9];
  const float* Wo = (const float*)d_in[10];
  const float* bo = (const float*)d_in[11];

  char* ws = (char*)d_ws;
  const size_t fsz = (size_t)BDIM * SEQ * DMODEL * sizeof(_Float16);  // 8 MB
  _Float16* Qh = (_Float16*)(ws + 0 * fsz);
  _Float16* Kh = (_Float16*)(ws + 1 * fsz);
  _Float16* Vt = (_Float16*)(ws + 2 * fsz);
  _Float16* At = (_Float16*)(ws + 3 * fsz);

  float* out     = (float*)d_out;
  float* weights = out + (size_t)BDIM * SEQ * DMODEL;

  dim3 g1(BDIM * SEQ / 128, DMODEL / 64, 3);
  proj_qkv_kernel<<<g1, 256, 0, stream>>>(query, key, value,
                                          Wq, bq, Wk, bk, Wv, bv,
                                          Qh, Kh, Vt);

  dim3 g2(SEQ / 64, NHEAD, BDIM);
  attention_kernel<<<g2, 128, 0, stream>>>(Qh, Kh, Vt, mask, weights, At);

  dim3 g3(BDIM * SEQ / 128, DMODEL / 64, 1);
  proj_out_kernel<<<g3, 256, 0, stream>>>(At, Wo, bo, out);
}
